// Encoder_23192823398615
// MI455X (gfx1250) — matrix operand order repaired and checked by reading, not compiled, not run on hardware
//
#include <hip/hip_runtime.h>

typedef __attribute__((ext_vector_type(16))) _Float16 v16h;
typedef __attribute__((ext_vector_type(8)))  _Float16 v8h;
typedef __attribute__((ext_vector_type(8)))  float    v8f;

#define LDSTRIDE 264   // halves per LDS row: 256 cols + 8 pad (row stride 528B, 16B aligned)

// packed-weight offsets in halves (fragment-major f16 layout)
#define OFF_P0 0        // 32(pad)x64
#define OFF_P1 2048     // 64x128
#define OFF_P2 10240    // 128x256
#define OFF_P3 43008    // 256x256
#define OFF_L0 108544   // 256x256
#define OFF_L1 174080   // 256x256
#define OFF_MW 239616   // 256x256
#define OFF_VW 305152   // 256x256
#define LAT_OFF_BYTES (768*1024)   // f32 latent [4096,256] lives here in d_ws

__device__ __forceinline__ float lrelu(float v){ return fmaxf(v, 0.01f*v); }

__device__ __forceinline__ v8f wmma16(v16h a, v16h b, v8f c){
  return __builtin_amdgcn_wmma_f32_16x16x32_f16(false, a, false, b, (short)0, c,
                                                false, false);
}

// A-fragment (16x32 f16) from LDS row-major activations.
// Lane layout (ISA 7.12.2): lanes 0-15 hold M=0..15, K in {g*8..g*8+7, 16+g*8..}.
__device__ __forceinline__ v16h load_a(const _Float16* in, int row, int kt, int g){
  const _Float16* p = in + row*LDSTRIDE + kt*32 + g*8;
  v8h lo = *(const v8h*)p;          // K = kt*32 + g*8 + 0..7
  v8h hi = *(const v8h*)(p + 16);   // K = kt*32 + 16 + g*8 + 0..7
  v16h a;
#pragma unroll
  for (int i = 0; i < 8; ++i){ a[i] = lo[i]; a[i+8] = hi[i]; }
  return a;
}

// One fused Linear(+LeakyReLU) layer: 256 rows (8 waves x 32 rows), LDS->LDS.
// N blocked by 4 tiles: A-fragments loaded once per kt feed 8 WMMAs.
template<int K, int NOUT, bool RELU>
__device__ __forceinline__ void mlp_layer(const _Float16* __restrict__ wp,
                                          const float* __restrict__ bias,
                                          const _Float16* in, _Float16* out,
                                          int wave, int lane){
  const int g  = lane >> 4;
  const int nl = lane & 15;
  const int rb = wave * 32;
#pragma unroll 1
  for (int ng = 0; ng < NOUT/64; ++ng){
    v8f acc[4][2];
#pragma unroll
    for (int j = 0; j < 4; ++j){ acc[j][0] = (v8f){}; acc[j][1] = (v8f){}; }
    const _Float16* wbase = wp + (size_t)(ng*4)*(K/32)*512 + lane*16;
#pragma unroll
    for (int kt = 0; kt < K/32; ++kt){
      v16h a0 = load_a(in, rb + nl,      kt, g);
      v16h a1 = load_a(in, rb + 16 + nl, kt, g);
#pragma unroll
      for (int j = 0; j < 4; ++j){
        v16h b = *(const v16h*)(wbase + ((size_t)j*(K/32) + kt)*512);
        acc[j][0] = wmma16(a0, b, acc[j][0]);
        acc[j][1] = wmma16(a1, b, acc[j][1]);
      }
      __builtin_prefetch(wbase + (size_t)(kt+1)*512, 0, 1);
    }
#pragma unroll
    for (int j = 0; j < 4; ++j){
      const int nc = (ng*4 + j)*16 + nl;
      const float bv = bias[nc];
#pragma unroll
      for (int r = 0; r < 8; ++r){
        float v0 = acc[j][0][r] + bv, v1 = acc[j][1][r] + bv;
        if (RELU){ v0 = lrelu(v0); v1 = lrelu(v1); }
        out[(rb      + r + 8*g)*LDSTRIDE + nc] = (_Float16)v0;
        out[(rb + 16 + r + 8*g)*LDSTRIDE + nc] = (_Float16)v1;
      }
    }
  }
}

// Linear head writing f32 to global (no activation), same N-blocking.
template<int K>
__device__ __forceinline__ void head_layer(const _Float16* __restrict__ wp,
                                           const float* __restrict__ bias,
                                           const _Float16* in,
                                           float* __restrict__ gout,  // ld = 256
                                           int wave, int lane){
  const int g  = lane >> 4;
  const int nl = lane & 15;
  const int rb = wave * 32;
#pragma unroll 1
  for (int ng = 0; ng < 4; ++ng){
    v8f acc[4][2];
#pragma unroll
    for (int j = 0; j < 4; ++j){ acc[j][0] = (v8f){}; acc[j][1] = (v8f){}; }
    const _Float16* wbase = wp + (size_t)(ng*4)*(K/32)*512 + lane*16;
#pragma unroll
    for (int kt = 0; kt < K/32; ++kt){
      v16h a0 = load_a(in, rb + nl,      kt, g);
      v16h a1 = load_a(in, rb + 16 + nl, kt, g);
#pragma unroll
      for (int j = 0; j < 4; ++j){
        v16h b = *(const v16h*)(wbase + ((size_t)j*(K/32) + kt)*512);
        acc[j][0] = wmma16(a0, b, acc[j][0]);
        acc[j][1] = wmma16(a1, b, acc[j][1]);
      }
    }
#pragma unroll
    for (int j = 0; j < 4; ++j){
      const int nc = (ng*4 + j)*16 + nl;
      const float bv = bias[nc];
#pragma unroll
      for (int r = 0; r < 8; ++r){
        gout[(rb      + r + 8*g)*256 + nc] = acc[j][0][r] + bv;
        gout[(rb + 16 + r + 8*g)*256 + nc] = acc[j][1][r] + bv;
      }
    }
  }
}

// Pack fp32 weight [NOUT][KREAL] into WMMA-B fragment order (f16), K padded to KPAD.
// B layout (ISA 7.12): within a 32-K fragment, lane half h holds K=8h+t for t<8,
// and K=16+8h+(t-8) for t>=8; N = lane%16.
template<int KPAD, int KREAL, int NOUT>
__global__ void pack_w(const float* __restrict__ w, _Float16* __restrict__ dst){
  int e = blockIdx.x*256 + threadIdx.x;
  if (e >= NOUT*KPAD) return;
  int f = e >> 9;
  int l = (e >> 4) & 31;
  int t = e & 15;
  int nt = f / (KPAD/32);
  int kt = f % (KPAD/32);
  int n  = nt*16 + (l & 15);
  int k  = kt*32 + ((t >> 3) << 4) + ((l >> 4) << 3) + (t & 7);
  float v = (k < KREAL) ? w[n*KREAL + k] : 0.0f;
  dst[e] = (_Float16)v;
}

// ---------------- Kernel 1: fused point MLP + segment-mean ----------------
__global__ __launch_bounds__(256) void point_encoder(
    const float* __restrict__ points, const int* __restrict__ idx,
    const _Float16* __restrict__ wp,
    const float* __restrict__ pb0, const float* __restrict__ pb1,
    const float* __restrict__ pb2, const float* __restrict__ pb3,
    float* __restrict__ latent){
  extern __shared__ _Float16 smem[];
  _Float16* bufA   = smem;
  _Float16* bufB   = smem + 256*LDSTRIDE;
  float*    latacc = (float*)(smem + 2*256*LDSTRIDE);

  const int b     = blockIdx.x;
  const int start = (b == 0) ? 0 : idx[b-1];
  const int cnt   = idx[b] - start;
  const int lane  = threadIdx.x & 31;
  const int wave  = threadIdx.x >> 5;
  const int g     = lane >> 4;
  const int nl    = lane & 15;
  const int rb    = wave * 32;

  latacc[threadIdx.x] = 0.0f;

  // ---- layer 0: points [cnt x 4] (K zero-padded to 32) -> bufA [256 x 64]
  v16h a0 = {}, a1 = {};
  if (g == 0){
    const int r0 = rb + nl, r1 = rb + 16 + nl;
    if (r0 < cnt){
      float4 p = ((const float4*)points)[start + r0];
      a0[0] = (_Float16)p.x; a0[1] = (_Float16)p.y;
      a0[2] = (_Float16)p.z; a0[3] = (_Float16)p.w;
    }
    if (r1 < cnt){
      float4 p = ((const float4*)points)[start + r1];
      a1[0] = (_Float16)p.x; a1[1] = (_Float16)p.y;
      a1[2] = (_Float16)p.z; a1[3] = (_Float16)p.w;
    }
  }
#pragma unroll
  for (int nt = 0; nt < 4; ++nt){
    v16h bfrag = *(const v16h*)(wp + OFF_P0 + nt*512 + lane*16);
    v8f c0 = {}, c1 = {};
    c0 = wmma16(a0, bfrag, c0);
    c1 = wmma16(a1, bfrag, c1);
    const float bv = pb0[nt*16 + nl];
#pragma unroll
    for (int r = 0; r < 8; ++r){
      bufA[(rb      + r + 8*g)*LDSTRIDE + nt*16 + nl] = (_Float16)lrelu(c0[r] + bv);
      bufA[(rb + 16 + r + 8*g)*LDSTRIDE + nt*16 + nl] = (_Float16)lrelu(c1[r] + bv);
    }
  }
  __syncthreads();

  mlp_layer< 64, 128, true>(wp + OFF_P1, pb1, bufA, bufB, wave, lane);
  __syncthreads();
  mlp_layer<128, 256, true>(wp + OFF_P2, pb2, bufB, bufA, wave, lane);
  __syncthreads();

  // ---- layer 3 (256->256, no activation) fused with segment reduction.
  // Bias is row-uniform so it is added once after the mean.
#pragma unroll 1
  for (int ng = 0; ng < 4; ++ng){
    v8f acc[4][2];
#pragma unroll
    for (int j = 0; j < 4; ++j){ acc[j][0] = (v8f){}; acc[j][1] = (v8f){}; }
    const _Float16* wbase = wp + OFF_P3 + (size_t)(ng*4)*8*512 + lane*16;
#pragma unroll
    for (int kt = 0; kt < 8; ++kt){
      v16h f0 = load_a(bufA, rb + nl,      kt, g);
      v16h f1 = load_a(bufA, rb + 16 + nl, kt, g);
#pragma unroll
      for (int j = 0; j < 4; ++j){
        v16h bb = *(const v16h*)(wbase + (size_t)(j*8 + kt)*512);
        acc[j][0] = wmma16(f0, bb, acc[j][0]);
        acc[j][1] = wmma16(f1, bb, acc[j][1]);
      }
    }
#pragma unroll
    for (int j = 0; j < 4; ++j){
      float s = 0.0f;
#pragma unroll
      for (int r = 0; r < 8; ++r){
        if (rb      + r + 8*g < cnt) s += acc[j][0][r];
        if (rb + 16 + r + 8*g < cnt) s += acc[j][1][r];
      }
      s += __shfl_xor(s, 16, 32);           // combine the two half-wave row groups
      if (g == 0) atomicAdd(&latacc[(ng*4 + j)*16 + nl], s);  // ds_add_f32 across waves
    }
  }
  __syncthreads();

  latent[b*256 + threadIdx.x] = latacc[threadIdx.x] / (float)cnt + pb3[threadIdx.x];
}

// ---------------- Kernel 2: latent MLP + mu / logvar heads ----------------
__global__ __launch_bounds__(256) void latent_encoder(
    const float* __restrict__ latent, const _Float16* __restrict__ wp,
    const float* __restrict__ lb0, const float* __restrict__ lb1,
    const float* __restrict__ mb,  const float* __restrict__ vb,
    float* __restrict__ out){
  extern __shared__ _Float16 smem[];
  _Float16* bufA = smem;
  _Float16* bufB = smem + 256*LDSTRIDE;
  const int lane = threadIdx.x & 31;
  const int wave = threadIdx.x >> 5;
  const int row0 = blockIdx.x * 256;

  // stage 256 rows of f32 latent as f16 activations
  for (int i = threadIdx.x; i < 256*256; i += 256){
    int r = i >> 8, c = i & 255;
    bufA[r*LDSTRIDE + c] = (_Float16)latent[(size_t)(row0 + r)*256 + c];
  }
  __syncthreads();

  mlp_layer<256, 256, true>(wp + OFF_L0, lb0, bufA, bufB, wave, lane);
  __syncthreads();
  mlp_layer<256, 256, true>(wp + OFF_L1, lb1, bufB, bufA, wave, lane);
  __syncthreads();

  head_layer<256>(wp + OFF_MW, mb, bufA, out + (size_t)row0*256,                    wave, lane);
  head_layer<256>(wp + OFF_VW, vb, bufA, out + (size_t)4096*256 + (size_t)row0*256, wave, lane);
}

extern "C" void kernel_launch(void* const* d_in, const int* in_sizes, int n_in,
                              void* d_out, int out_size, void* d_ws, size_t ws_size,
                              hipStream_t stream) {
  const float* points = (const float*)d_in[0];
  const int*   idx    = (const int*)  d_in[1];
  const float* pw0 = (const float*)d_in[2];  const float* pb0 = (const float*)d_in[3];
  const float* pw1 = (const float*)d_in[4];  const float* pb1 = (const float*)d_in[5];
  const float* pw2 = (const float*)d_in[6];  const float* pb2 = (const float*)d_in[7];
  const float* pw3 = (const float*)d_in[8];  const float* pb3 = (const float*)d_in[9];
  const float* lw0 = (const float*)d_in[10]; const float* lb0 = (const float*)d_in[11];
  const float* lw1 = (const float*)d_in[12]; const float* lb1 = (const float*)d_in[13];
  const float* mw  = (const float*)d_in[14]; const float* mb  = (const float*)d_in[15];
  const float* vw  = (const float*)d_in[16]; const float* vb  = (const float*)d_in[17];

  _Float16* wp     = (_Float16*)d_ws;
  float*    latent = (float*)((char*)d_ws + LAT_OFF_BYTES);
  float*    out    = (float*)d_out;

  // ---- pack weights fp32 -> f16 WMMA-B fragments
  pack_w< 32,   4,  64><<<(  32*64 + 255)/256, 256, 0, stream>>>(pw0, wp + OFF_P0);
  pack_w< 64,  64, 128><<<( 64*128 + 255)/256, 256, 0, stream>>>(pw1, wp + OFF_P1);
  pack_w<128, 128, 256><<<(128*256 + 255)/256, 256, 0, stream>>>(pw2, wp + OFF_P2);
  pack_w<256, 256, 256><<<(256*256 + 255)/256, 256, 0, stream>>>(pw3, wp + OFF_P3);
  pack_w<256, 256, 256><<<(256*256 + 255)/256, 256, 0, stream>>>(lw0, wp + OFF_L0);
  pack_w<256, 256, 256><<<(256*256 + 255)/256, 256, 0, stream>>>(lw1, wp + OFF_L1);
  pack_w<256, 256, 256><<<(256*256 + 255)/256, 256, 0, stream>>>(mw,  wp + OFF_MW);
  pack_w<256, 256, 256><<<(256*256 + 255)/256, 256, 0, stream>>>(vw,  wp + OFF_VW);

  // ---- fused point MLP + segment mean: one WG per cloud, 270 KB LDS (CDNA5 WGP has 320 KB)
  size_t shm1 = (size_t)2*256*LDSTRIDE*sizeof(_Float16) + 256*sizeof(float);
  point_encoder<<<4096, 256, shm1, stream>>>(points, idx, wp, pb0, pb1, pb2, pb3, latent);

  // ---- latent MLP + heads
  size_t shm2 = (size_t)2*256*LDSTRIDE*sizeof(_Float16);
  latent_encoder<<<16, 256, shm2, stream>>>(latent, wp, lb0, lb1, mb, vb, out);
}
